// EfficientMultiHeadAttention_38946763440967
// MI455X (gfx1250) — matrix operand order repaired and checked by reading, not compiled, not run on hardware
//
#include <hip/hip_runtime.h>

typedef __attribute__((ext_vector_type(16))) __bf16 bf16x16;
typedef __attribute__((ext_vector_type(8)))  float  f32x8;
typedef __attribute__((ext_vector_type(4)))  int    i32x4;

constexpr int NB = 4;
constexpr int NT = 2048;
constexpr int NE = 1024;
constexpr int NH = 16;
constexpr int ND = 64;
constexpr int NR = NB * NT;          // 8192 token rows
constexpr float SCALE = 0.125f;      // 1/sqrt(64), folded into Q

#if defined(__HIP_DEVICE_COMPILE__) && defined(__has_builtin)
#if __has_builtin(__builtin_amdgcn_global_load_async_to_lds_b128) && \
    __has_builtin(__builtin_amdgcn_s_wait_asynccnt)
#define HAVE_ASYNC_LDS 1
#endif
#endif

__device__ __forceinline__ f32x8 wmma_bf16(bf16x16 a, bf16x16 b, f32x8 c) {
  // D = A(16x32 bf16) * B(32x16 bf16) + C(16x16 f32)
  return __builtin_amdgcn_wmma_f32_16x16x32_bf16(false, a, false, b, (short)0, c,
                                                 false, false);
}

// ---------------- prep: fp32 -> bf16 (and weight transpose) ----------------
__global__ void k_cvt_bf16(const float* __restrict__ in, __bf16* __restrict__ out,
                           int n) {
  int i = blockIdx.x * blockDim.x + threadIdx.x;
  if (i < n) out[i] = (__bf16)in[i];
}

// in: [K][N] row-major fp32  ->  out: [N][K] bf16 (WMMA B-fragments then load
// contiguously along K)
__global__ void k_transpose_bf16(const float* __restrict__ in,
                                 __bf16* __restrict__ out, int K, int N) {
  int i = blockIdx.x * blockDim.x + threadIdx.x;
  if (i < K * N) {
    int kk = i / N, n = i - kk * N;
    out[(size_t)n * K + kk] = (__bf16)in[i];
  }
}

// ---------------- QKV projection: one wave computes a 16x64 tile (A reused
// across 4 column tiles). Q scaled by 1/sqrt(D); V stored transposed
// (Vt[bh][d][t]) so P@V B-fragments later load contiguously.
__global__ __launch_bounds__(256) void k_qkv(
    const __bf16* __restrict__ xb, const __bf16* __restrict__ wT,
    const float* __restrict__ bias, __bf16* __restrict__ q,
    __bf16* __restrict__ kbuf, __bf16* __restrict__ vt) {
  const int wave = threadIdx.x >> 5;
  const int lane = threadIdx.x & 31;
  const int tile = blockIdx.x * 8 + wave;            // 16x64 tile per wave
  const int ntc  = (3 * NE) / 64;                    // 48 column groups
  const int ti = tile / ntc, tc = tile % ntc;
  const int i0 = ti * 16, c0 = tc * 64;
  const int r16 = lane & 15, half = lane >> 4;

  const __bf16* arow = xb + (size_t)(i0 + r16) * NE + half * 16;
  f32x8 acc[4] = {};
  for (int k0 = 0; k0 < NE; k0 += 32) {
    bf16x16 a;
#pragma unroll
    for (int j = 0; j < 16; ++j) { a[j] = arow[k0 + j]; }
    __builtin_prefetch(arow + k0 + 256, 0, 0);
#pragma unroll
    for (int g = 0; g < 4; ++g) {
      const __bf16* brow = wT + (size_t)(c0 + g * 16 + r16) * NE + half * 16 + k0;
      bf16x16 b;
#pragma unroll
      for (int j = 0; j < 16; ++j) b[j] = brow[j];
      acc[g] = wmma_bf16(a, b, acc[g]);
    }
  }
  // scatter: col -> (which, head, d)
#pragma unroll
  for (int g = 0; g < 4; ++g) {
    const int col = c0 + g * 16 + r16;
    const int which = col >> 10;
    const int h = (col & 1023) >> 6;
    const int d = col & 63;
    const float scl = (which == 0) ? SCALE : 1.0f;
    const float bia = bias[col];
#pragma unroll
    for (int r = 0; r < 8; ++r) {
      int t = i0 + half * 8 + r;
      int bidx = t >> 11;               // / 2048
      int tt = t & 2047;
      const __bf16 val = (__bf16)((acc[g][r] + bia) * scl);
      if (which == 2)                    // V transposed: [bh][d][t]
        vt[(((size_t)(bidx * NH + h)) * ND + d) * NT + tt] = val;
      else {
        __bf16* dst = (which == 0) ? q : kbuf;   // [bh][t][d]
        dst[(((size_t)(bidx * NH + h)) * NT + tt) * ND + d] = val;
      }
    }
  }
}

// ---------------- fused attention: one WG owns (b,h) x 16 rows, full T
// scores (WMMA, K tiles staged via async-to-LDS when available) -> LDS strip
// -> softmax in LDS -> attn written once -> P@V (WMMA, contiguous Vt loads)
__global__ __launch_bounds__(256) void k_attn(
    const __bf16* __restrict__ q, const __bf16* __restrict__ k,
    const __bf16* __restrict__ vt, float* __restrict__ attn,
    __bf16* __restrict__ ctx) {
  __shared__ float strip[16 * NT];      // 128 KB: 16 rows x 2048 scores
  __shared__ float red[16 * 16];
  __shared__ float ctxacc[16 * ND];     // 4 KB
#ifdef HAVE_ASYNC_LDS
  __shared__ __align__(16) __bf16 ktile[8][16 * ND];  // 2 KB per wave
#endif
  const int itile = blockIdx.x & 127;
  const int bh = blockIdx.x >> 7;       // b*16 + h
  const int i0 = itile * 16;
  const int wave = threadIdx.x >> 5, lane = threadIdx.x & 31;
  const int r16 = lane & 15, half = lane >> 4;
  const __bf16* Qp = q  + (size_t)bh * NT * ND;
  const __bf16* Kp = k  + (size_t)bh * NT * ND;
  const __bf16* Vt = vt + (size_t)bh * ND * NT;

#pragma unroll
  for (int u = 0; u < 4; ++u) ctxacc[threadIdx.x * 4 + u] = 0.0f;

  // Q A-fragments for the two K=32 steps over D=64 (held in regs, reused)
  bf16x16 a0, a1;
  {
    const __bf16* qrow = Qp + (size_t)(i0 + r16) * ND + half * 16;
#pragma unroll
    for (int j = 0; j < 16; ++j) {
      a0[j] = qrow[j];
      a1[j] = qrow[32 + j];
    }
  }
  // scores: waves split the 128 j-tiles
  for (int jt = wave; jt < NT / 16; jt += 8) {
    const int j0 = jt * 16;
    bf16x16 b0, b1;
#ifdef HAVE_ASYNC_LDS
    {
      // stage the 16x64 bf16 K tile (2 KB) into this wave's LDS buffer:
      // 4 async b128 issues x 32 lanes x 16 B
      const __bf16* gbase = Kp + (size_t)j0 * ND + (size_t)r16 * ND + half * 8;
      __bf16* lbase = &ktile[wave][r16 * ND + half * 8];
#pragma unroll
      for (int chunk = 0; chunk < 4; ++chunk) {
        __builtin_amdgcn_global_load_async_to_lds_b128(
            (__attribute__((address_space(1))) i32x4*)(gbase + chunk * 16),
            (__attribute__((address_space(3))) i32x4*)(lbase + chunk * 16), 0, 0);
      }
      __builtin_amdgcn_s_wait_asynccnt(0);
      const __bf16* kt = &ktile[wave][r16 * ND + half * 16];
#pragma unroll
      for (int j = 0; j < 16; ++j) { b0[j] = kt[j]; b1[j] = kt[32 + j]; }
    }
#else
    {
      const __bf16* krow = Kp + (size_t)(j0 + r16) * ND + half * 16;
#pragma unroll
      for (int j = 0; j < 16; ++j) { b0[j] = krow[j]; b1[j] = krow[32 + j]; }
    }
#endif
    f32x8 c = {};
    c = wmma_bf16(a0, b0, c);
    c = wmma_bf16(a1, b1, c);
#pragma unroll
    for (int r = 0; r < 8; ++r) strip[(half * 8 + r) * NT + j0 + r16] = c[r];
  }
  __syncthreads();

  // softmax: 16 threads per row, 128 cols each
  const int row = threadIdx.x >> 4, sub = threadIdx.x & 15;
  float* srow = strip + row * NT;
  const int cb = sub * 128;
  float mx = -3.0e38f;
  for (int c2 = 0; c2 < 128; ++c2) mx = fmaxf(mx, srow[cb + c2]);
  red[row * 16 + sub] = mx;
  __syncthreads();
  if (sub == 0) {
    float m = -3.0e38f;
    for (int s = 0; s < 16; ++s) m = fmaxf(m, red[row * 16 + s]);
    red[row * 16] = m;
  }
  __syncthreads();
  const float rmax = red[row * 16];
  float sum = 0.0f;
  for (int c2 = 0; c2 < 128; ++c2) {
    float e = __expf(srow[cb + c2] - rmax);
    srow[cb + c2] = e;
    sum += e;
  }
  __syncthreads();
  red[row * 16 + sub] = sum;
  __syncthreads();
  if (sub == 0) {
    float t = 0.0f;
    for (int s = 0; s < 16; ++s) t += red[row * 16 + s];
    red[row * 16] = t;
  }
  __syncthreads();
  const float inv = 1.0f / red[row * 16];
  float* arow = attn + ((size_t)bh * NT + i0 + row) * NT;
  for (int c2 = 0; c2 < 128; ++c2) {
    float p = srow[cb + c2] * inv;
    srow[cb + c2] = p;
    arow[cb + c2] = p;                 // attn written exactly once
  }
  __syncthreads();

  // ctx = P @ V ; K=2048 -> 64 ksteps of 32, waves split K, reduce via LDS.
  // Vt[bh][d][t] makes every B-fragment a pair of contiguous 32B loads.
  f32x8 acc[4] = {};
  for (int ks = wave; ks < NT / 32; ks += 8) {
    const int kk0 = ks * 32;
    bf16x16 a;
    const float* prow = strip + r16 * NT + kk0 + half * 16;
#pragma unroll
    for (int j = 0; j < 16; ++j) { a[j] = (__bf16)prow[j]; }
#pragma unroll
    for (int nt = 0; nt < 4; ++nt) {
      bf16x16 b;
      const __bf16* vp = Vt + ((size_t)nt * 16 + r16) * NT + kk0 + half * 16;
#pragma unroll
      for (int j = 0; j < 16; ++j) b[j] = vp[j];
      acc[nt] = wmma_bf16(a, b, acc[nt]);
    }
  }
#pragma unroll
  for (int nt = 0; nt < 4; ++nt)
#pragma unroll
    for (int r = 0; r < 8; ++r)
      atomicAdd(&ctxacc[(half * 8 + r) * ND + nt * 16 + r16], acc[nt][r]);
  __syncthreads();

  const int bb = bh >> 4, hh = bh & 15;
#pragma unroll
  for (int u = 0; u < 4; ++u) {
    int idx = threadIdx.x * 4 + u;
    int m = idx >> 6, d = idx & 63;
    ctx[((size_t)(bb * NT + i0 + m)) * NE + hh * ND + d] = (__bf16)ctxacc[idx];
  }
}

// ---------------- out proj + residual + LayerNorm, fused per 16-row strip
__global__ __launch_bounds__(256) void k_out(
    const __bf16* __restrict__ ctx, const __bf16* __restrict__ woT,
    const float* __restrict__ ob, const float* __restrict__ x,
    const float* __restrict__ g, const float* __restrict__ bln,
    float* __restrict__ y) {
  __shared__ float strip[16 * NE];     // 64 KB
  __shared__ float red[16 * 16];
  const int i0 = blockIdx.x * 16;
  const int wave = threadIdx.x >> 5, lane = threadIdx.x & 31;
  const int r16 = lane & 15, half = lane >> 4;

  const __bf16* arow = ctx + (size_t)(i0 + r16) * NE + half * 16;
  f32x8 acc[8] = {};
  for (int k0 = 0; k0 < NE; k0 += 32) {
    bf16x16 a;
#pragma unroll
    for (int j = 0; j < 16; ++j) { a[j] = arow[k0 + j]; }
    __builtin_prefetch(arow + k0 + 256, 0, 0);
#pragma unroll
    for (int w8 = 0; w8 < 8; ++w8) {
      const int nt = wave + w8 * 8;
      const __bf16* brow = woT + (size_t)(nt * 16 + r16) * NE + half * 16 + k0;
      bf16x16 b;
#pragma unroll
      for (int j = 0; j < 16; ++j) b[j] = brow[j];
      acc[w8] = wmma_bf16(a, b, acc[w8]);
    }
  }
#pragma unroll
  for (int w8 = 0; w8 < 8; ++w8) {
    const int nt = wave + w8 * 8;
    const int col = nt * 16 + r16;
    const float bia = ob[col];
#pragma unroll
    for (int r = 0; r < 8; ++r) {
      const int m = half * 8 + r;
      strip[m * NE + col] = acc[w8][r] + bia + x[(size_t)(i0 + m) * NE + col];
    }
  }
  __syncthreads();

  const int row = threadIdx.x >> 4, sub = threadIdx.x & 15;
  float* srow = strip + row * NE;
  const int cb = sub * 64;
  float s = 0.0f;
  for (int c2 = 0; c2 < 64; ++c2) s += srow[cb + c2];
  red[row * 16 + sub] = s;
  __syncthreads();
  if (sub == 0) {
    float t = 0.0f;
    for (int u = 0; u < 16; ++u) t += red[row * 16 + u];
    red[row * 16] = t * (1.0f / NE);
  }
  __syncthreads();
  const float mu = red[row * 16];
  __syncthreads();
  float vs = 0.0f;
  for (int c2 = 0; c2 < 64; ++c2) { float dd = srow[cb + c2] - mu; vs += dd * dd; }
  red[row * 16 + sub] = vs;
  __syncthreads();
  if (sub == 0) {
    float t = 0.0f;
    for (int u = 0; u < 16; ++u) t += red[row * 16 + u];
    red[row * 16] = t * (1.0f / NE);
  }
  __syncthreads();
  const float rstd = rsqrtf(red[row * 16] + 1e-5f);
  for (int c2 = 0; c2 < 64; ++c2) {
    const int col = cb + c2;
    y[(size_t)(i0 + row) * NE + col] = (srow[col] - mu) * rstd * g[col] + bln[col];
  }
}

extern "C" void kernel_launch(void* const* d_in, const int* in_sizes, int n_in,
                              void* d_out, int out_size, void* d_ws, size_t ws_size,
                              hipStream_t stream) {
  const float* x     = (const float*)d_in[0];
  const float* qkv_w = (const float*)d_in[1];
  const float* qkv_b = (const float*)d_in[2];
  const float* out_w = (const float*)d_in[3];
  const float* out_b = (const float*)d_in[4];
  const float* ln_g  = (const float*)d_in[5];
  const float* ln_b  = (const float*)d_in[6];
  float* y    = (float*)d_out;
  float* attn = y + (size_t)NB * NT * NE;   // outputs concatenated: y then attn

  // workspace layout (bf16 staging buffers), ~92 MB total
  char* ws = (char*)d_ws;
  __bf16* xb    = (__bf16*)(ws);                       // 8192x1024   (16 MB)
  __bf16* wqkvT = (__bf16*)(ws + (size_t)16777216);    // 3072x1024   ( 6 MB)
  __bf16* woutT = (__bf16*)(ws + (size_t)23068672);    // 1024x1024   ( 2 MB)
  __bf16* qb    = (__bf16*)(ws + (size_t)25165824);    // [B*H][T][D] (16 MB)
  __bf16* kb    = (__bf16*)(ws + (size_t)41943040);    // [B*H][T][D] (16 MB)
  __bf16* vtb   = (__bf16*)(ws + (size_t)58720256);    // [B*H][D][T] (16 MB)
  __bf16* ctxb  = (__bf16*)(ws + (size_t)75497472);    // 8192x1024   (16 MB)

  k_cvt_bf16<<<(NR * NE) / 256, 256, 0, stream>>>(x, xb, NR * NE);
  k_transpose_bf16<<<(NE * 3 * NE) / 256, 256, 0, stream>>>(qkv_w, wqkvT, NE, 3 * NE);
  k_transpose_bf16<<<(NE * NE) / 256, 256, 0, stream>>>(out_w, woutT, NE, NE);
  k_qkv<<<((NR / 16) * ((3 * NE) / 64)) / 8, 256, 0, stream>>>(xb, wqkvT, qkv_b,
                                                               qb, kb, vtb);
  k_attn<<<NB * NH * (NT / 16), 256, 0, stream>>>(qb, kb, vtb, attn, ctxb);
  k_out<<<NR / 16, 256, 0, stream>>>(ctxb, woutT, out_b, x, ln_g, ln_b, y);
}
